// MAABlock_37520834297980
// MI455X (gfx1250) — matrix operand order repaired and checked by reading, not compiled, not run on hardware
//
#include <hip/hip_runtime.h>

typedef __attribute__((ext_vector_type(16))) _Float16 v16h;
typedef __attribute__((ext_vector_type(8)))  _Float16 v8h;
typedef __attribute__((ext_vector_type(8)))  float    v8f;

union V16 { v16h v; v8h h[2]; };

#define WMMA_F16(A_, B_, C_) \
  __builtin_amdgcn_wmma_f32_16x16x32_f16(false, (A_), false, (B_), (short)0, (C_), false, false)

// ---------------------------------------------------------------------------
// Index helpers for the dilated 8x8 blocking:
//   blocked row r = b*4096 + m*64 + n,  m=(hb,wb), n=(a,bx)
//   token  row t = b*4096 + (a*8+hb)*64 + (bx*8+wb)
// ---------------------------------------------------------------------------
static __device__ __forceinline__ int blocked_to_token(int r) {
  int b  = r >> 12;
  int m  = (r >> 6) & 63;
  int n  = r & 63;
  int hb = m >> 3, wb = m & 7;
  int a  = n >> 3, bx = n & 7;
  return (b << 12) + (a * 8 + hb) * 64 + (bx * 8 + wb);
}
static __device__ __forceinline__ int token_to_blocked(int t) {
  int b  = t >> 12;
  int p  = t & 4095;
  int h  = p >> 6, w = p & 63;
  int a  = h >> 3, hb = h & 7;
  int bx = w >> 3, wb = w & 7;
  return (b << 12) + ((hb * 8 + wb) << 6) + (a * 8 + bx);
}

// ---------------------------------------------------------------------------
// Weight packing: fp32 -> f16 in the WMMA B-operand tile layout.
// For logical B[K][N]: lane = (kr>>4)*16 | (n&15), element i = kr&15,
// tile index (kt, nt). One fragment = 32 lanes * 16 contiguous halves.
// ---------------------------------------------------------------------------
__global__ void pack_weights(const float* __restrict__ q, const float* __restrict__ k,
                             const float* __restrict__ v, const float* __restrict__ o,
                             const float* __restrict__ w1, const float* __restrict__ w2,
                             _Float16* __restrict__ wqkv, _Float16* __restrict__ wo,
                             _Float16* __restrict__ w1p, _Float16* __restrict__ w2p,
                             int mode)
{
  int idx = blockIdx.x * 256 + threadIdx.x;
  int K, N;
  if (mode == 0)      { K = 256;  N = 2560; }
  else if (mode == 1) { K = 2048; N = 256;  }
  else                { K = 256;  N = 256;  }
  if (idx >= K * N) return;
  int kk = idx / N, n = idx - kk * N;
  float val;
  _Float16* dst;
  if (mode == 0) {
    if (n < 2048)       val = q[(n >> 8) * 65536 + kk * 256 + (n & 255)];
    else if (n < 2304)  val = k[kk * 256 + (n - 2048)];
    else                val = v[kk * 256 + (n - 2304)];
    dst = wqkv;
  } else if (mode == 1) {
    val = o[(kk >> 8) * 65536 + (kk & 255) * 256 + n];
    dst = wo;
  } else if (mode == 2) { val = w1[kk * 256 + n]; dst = w1p; }
  else                  { val = w2[kk * 256 + n]; dst = w2p; }
  int kt = kk >> 5, kr = kk & 31;
  int lane = ((kr >> 4) << 4) | (n & 15);
  int i = kr & 15;
  int nt = n >> 4;
  size_t p = ((size_t)(kt * (N >> 4) + nt) * 32 + (size_t)lane) * 16 + (size_t)i;
  dst[p] = (_Float16)val;
}

// ---------------------------------------------------------------------------
// LayerNorm (fp32 in, f16 out). One wave per token; lane owns 8 channels.
// blocked=1 -> write rows in blocked order (for attention path).
// ---------------------------------------------------------------------------
__global__ void ln_kernel(const float* __restrict__ x, const float* __restrict__ w,
                          const float* __restrict__ b, _Float16* __restrict__ out,
                          int blocked)
{
  int lane  = threadIdx.x & 31;
  int token = blockIdx.x * 8 + (threadIdx.x >> 5);
  const float* row = x + (size_t)token * 256;
  const float4* p = (const float4*)(row + lane * 8);
  float4 f0 = p[0], f1 = p[1];
  float vals[8] = { f0.x, f0.y, f0.z, f0.w, f1.x, f1.y, f1.z, f1.w };

  float s = 0.f;
  #pragma unroll
  for (int j = 0; j < 8; ++j) s += vals[j];
  #pragma unroll
  for (int off = 16; off >= 1; off >>= 1) s += __shfl_xor(s, off, 32);
  float mean = s * (1.0f / 256.0f);

  float vs = 0.f;
  #pragma unroll
  for (int j = 0; j < 8; ++j) { float d = vals[j] - mean; vs += d * d; }
  #pragma unroll
  for (int off = 16; off >= 1; off >>= 1) vs += __shfl_xor(vs, off, 32);
  float inv = rsqrtf(vs * (1.0f / 256.0f) + 1e-5f);

  int orow = blocked ? token_to_blocked(token) : token;
  int c0 = lane * 8;
  v8h ov;
  #pragma unroll
  for (int j = 0; j < 8; ++j)
    ov[j] = (_Float16)((vals[j] - mean) * inv * w[c0 + j] + b[c0 + j]);
  *(v8h*)(out + (size_t)orow * 256 + c0) = ov;
}

// ---------------------------------------------------------------------------
// f16 WMMA GEMM, 32x64 output block per wave (2 m-tiles x 4 n-tiles).
// A fragments reused across 4 n-tiles, B fragments across 2 m-tiles:
// 8 WMMAs per 12 b128 loads, A traffic /4, B traffic /2 vs 16x16/wave.
// A: row-major f16 (lda elems). Bp: packed B-operand layout.
// MODE 0: -> f16, ldo=2560 (QKV). MODE 1: row remap + x residual -> f32 (O-proj).
// MODE 2: +bias, relu -> f16 (MLP1). MODE 3: +bias + residual -> f32 (MLP2).
// ---------------------------------------------------------------------------
template<int MODE>
__global__ void gemm16(const _Float16* __restrict__ A, int lda,
                       const _Float16* __restrict__ Bp, int N, int K,
                       const float* __restrict__ bias,
                       const float* __restrict__ resid,
                       _Float16* __restrict__ outH, float* __restrict__ outF)
{
  int lane = threadIdx.x & 31;
  int wid  = blockIdx.x * 8 + (threadIdx.x >> 5);
  int nblk = N >> 6;                       // 64-wide column blocks
  int bm = wid / nblk;
  int bn = wid - bm * nblk;
  int r0 = bm << 5;                        // 32 rows per wave
  int c0 = bn << 6;                        // 64 cols per wave

  const _Float16* arow0 = A + (size_t)(r0 + (lane & 15)) * (size_t)lda;
  const _Float16* arow1 = arow0 + (size_t)16 * (size_t)lda;
  int ntiles = N >> 4;

  v8f acc[2][4] = {};
  for (int k0 = 0; k0 < K; k0 += 32) {
    int ak = k0 + ((lane >> 4) << 4);
    V16 a0, a1;
    a0.h[0] = *(const v8h*)(arow0 + ak);
    a0.h[1] = *(const v8h*)(arow0 + ak + 8);
    a1.h[0] = *(const v8h*)(arow1 + ak);
    a1.h[1] = *(const v8h*)(arow1 + ak + 8);
    size_t kbase = ((size_t)(k0 >> 5) * ntiles + (bn << 2)) * 32;
    #pragma unroll
    for (int j = 0; j < 4; ++j) {
      V16 bb;
      bb.v = *(const v16h*)(Bp + ((kbase + (size_t)j * 32) + lane) * 16);
      acc[0][j] = WMMA_F16(a0.v, bb.v, acc[0][j]);
      acc[1][j] = WMMA_F16(a1.v, bb.v, acc[1][j]);
    }
  }

  #pragma unroll
  for (int im = 0; im < 2; ++im) {
    int rbase = r0 + im * 16 + ((lane >> 4) << 3);
    #pragma unroll
    for (int j = 0; j < 4; ++j) {
      int c = c0 + j * 16 + (lane & 15);
      #pragma unroll
      for (int rr = 0; rr < 8; ++rr) {
        int r = rbase + rr;
        float vv = acc[im][j][rr];
        if constexpr (MODE == 0) {
          outH[(size_t)r * 2560 + c] = (_Float16)vv;
        } else if constexpr (MODE == 1) {
          int r2 = blocked_to_token(r);
          size_t oo = (size_t)r2 * 256 + c;
          outF[oo] = resid[oo] + vv;               // x + unblock(Z)
        } else if constexpr (MODE == 2) {
          vv += bias[c];
          vv = fmaxf(vv, 0.0f);
          outH[(size_t)r * 256 + c] = (_Float16)vv;
        } else {
          vv += bias[c];
          size_t oo = (size_t)r * 256 + c;
          outF[oo] = resid[oo] + vv;               // x1 + mlp
        }
      }
    }
  }
}

// ---------------------------------------------------------------------------
// Axial attention: one workgroup per (b, head, fixed index, branch) group.
// Row map in the blocked QKV matrix: row(i) = base + i*stride,
//   branch 0 (heads 0-3, fixed y): base = b*4096 + y, stride = 64
//   branch 1 (heads 4-7, fixed x): base = b*4096 + x*64, stride = 1
// QKV layout: [32768 rows x 2560], Q head h at col 256h, K at 2048, V at 2304.
// O overwrites Q in place (each Q entry read by exactly its own group).
// ---------------------------------------------------------------------------
__global__ void attn_kernel(_Float16* __restrict__ QKV)
{
  __shared__ __align__(32) _Float16 VT[256 * 64];   // V transposed [v][z]
  __shared__ __align__(16) float    SC[64 * 64];    // fp32 scores
  __shared__ __align__(16) _Float16 P[64 * 64];     // f16 probabilities

  int tid  = threadIdx.x;
  int lane = tid & 31;
  int wave = tid >> 5;

  int fixed  = blockIdx.x;          // 0..63
  int hh     = blockIdx.y;          // 0..3
  int branch = blockIdx.z >> 3;     // 0 / 1
  int b      = blockIdx.z & 7;

  int h      = branch ? (4 + hh) : hh;
  int hcol   = h << 8;
  int base   = (b << 12) + (branch ? (fixed << 6) : fixed);
  int stride = branch ? 1 : 64;

  // Stage V transposed: VT[v][z] = V[row(z)][v]
  for (int e = tid; e < 16384; e += 256) {
    int z = e >> 8, vv = e & 255;
    VT[vv * 64 + z] = QKV[(size_t)(base + z * stride) * 2560 + 2304 + vv];
  }
  __syncthreads();

  // Scores S[x][z] = Q_row(x) . K_row(z) over K=256 : 16 tiles, 2 per wave
  #pragma unroll
  for (int t2 = 0; t2 < 2; ++t2) {
    int tileid = wave * 2 + t2;
    int mi = tileid >> 2, ni = tileid & 3;
    size_t qrow = (size_t)(base + (mi * 16 + (lane & 15)) * stride) * 2560 + hcol;
    size_t krow = (size_t)(base + (ni * 16 + (lane & 15)) * stride) * 2560 + 2048;
    v8f acc = {};
    for (int k0 = 0; k0 < 256; k0 += 32) {
      int ak = k0 + ((lane >> 4) << 4);
      V16 a;
      a.h[0] = *(const v8h*)(QKV + qrow + ak);
      a.h[1] = *(const v8h*)(QKV + qrow + ak + 8);
      int bk = k0 + ((lane >> 4) << 4);
      V16 bb;
      bb.v = *(const v16h*)(QKV + krow + bk);
      acc = WMMA_F16(a.v, bb.v, acc);
    }
    int col = (ni << 4) + (lane & 15);
    int rb  = (mi << 4) + ((lane >> 4) << 3);
    #pragma unroll
    for (int rr = 0; rr < 8; ++rr) SC[(rb + rr) * 64 + col] = acc[rr];
  }
  __syncthreads();

  // Row softmax over z (64), probs -> f16
  if (tid < 64) {
    float mx = -3.0e38f;
    for (int z = 0; z < 64; ++z) mx = fmaxf(mx, SC[tid * 64 + z]);
    float s = 0.f;
    for (int z = 0; z < 64; ++z) {
      float e = __expf(SC[tid * 64 + z] - mx);
      s += e;
      SC[tid * 64 + z] = e;
    }
    float inv = 1.0f / s;
    for (int z = 0; z < 64; ++z) P[tid * 64 + z] = (_Float16)(SC[tid * 64 + z] * inv);
  }
  __syncthreads();

  // O[x][v] = P[x][z] * V[z][v] : M=64, N=256, K=64 -> 64 tiles, 8 per wave
  #pragma unroll
  for (int t2 = 0; t2 < 8; ++t2) {
    int idx = wave * 8 + t2;
    int mi = idx >> 4, ni = idx & 15;
    int ar = (mi << 4) + (lane & 15);
    int br = (ni << 4) + (lane & 15);
    v8f acc = {};
    #pragma unroll
    for (int k0 = 0; k0 < 64; k0 += 32) {
      int ak = k0 + ((lane >> 4) << 4);
      V16 a;
      a.h[0] = *(const v8h*)&P[ar * 64 + ak];
      a.h[1] = *(const v8h*)&P[ar * 64 + ak + 8];
      int bk = k0 + ((lane >> 4) << 4);
      V16 bb;
      bb.v = *(const v16h*)&VT[br * 64 + bk];
      acc = WMMA_F16(a.v, bb.v, acc);
    }
    int col = hcol + (ni << 4) + (lane & 15);
    int mb  = (mi << 4) + ((lane >> 4) << 3);
    #pragma unroll
    for (int rr = 0; rr < 8; ++rr) {
      size_t rowg = (size_t)(base + (mb + rr) * stride);
      QKV[rowg * 2560 + col] = (_Float16)acc[rr];
    }
  }
}

// ---------------------------------------------------------------------------
extern "C" void kernel_launch(void* const* d_in, const int* in_sizes, int n_in,
                              void* d_out, int out_size, void* d_ws, size_t ws_size,
                              hipStream_t stream)
{
  (void)in_sizes; (void)n_in; (void)out_size; (void)ws_size;
  const float* x    = (const float*)d_in[0];
  const float* ln1w = (const float*)d_in[1];
  const float* ln1b = (const float*)d_in[2];
  const float* qw   = (const float*)d_in[3];
  const float* kw   = (const float*)d_in[4];
  const float* vw   = (const float*)d_in[5];
  const float* ow   = (const float*)d_in[6];
  const float* ln2w = (const float*)d_in[7];
  const float* ln2b = (const float*)d_in[8];
  const float* w1   = (const float*)d_in[9];
  const float* b1   = (const float*)d_in[10];
  const float* w2   = (const float*)d_in[11];
  const float* b2   = (const float*)d_in[12];
  float* out = (float*)d_out;

  char* ws = (char*)d_ws;
  _Float16* QKV  = (_Float16*)(ws);                     // 32768 x 2560 f16 (160 MB)
  _Float16* A16  = (_Float16*)(ws + 167772160ull);      // 32768 x 256  f16
  float*    X1   = (float*)   (ws + 184549376ull);      // 32768 x 256  f32
  _Float16* H16  = (_Float16*)(ws + 218103808ull);      // 32768 x 256  f16
  _Float16* M16  = (_Float16*)(ws + 234881024ull);      // 32768 x 256  f16
  _Float16* WQKV = (_Float16*)(ws + 251658240ull);      // 256  x 2560  packed
  _Float16* WO   = (_Float16*)(ws + 252968960ull);      // 2048 x 256   packed
  _Float16* W1P  = (_Float16*)(ws + 254017536ull);      // 256  x 256   packed
  _Float16* W2P  = (_Float16*)(ws + 254148608ull);      // 256  x 256   packed

  // Pack weights to f16 WMMA B-layout
  pack_weights<<<2560, 256, 0, stream>>>(qw, kw, vw, ow, w1, w2, WQKV, WO, W1P, W2P, 0);
  pack_weights<<<2048, 256, 0, stream>>>(qw, kw, vw, ow, w1, w2, WQKV, WO, W1P, W2P, 1);
  pack_weights<<<256,  256, 0, stream>>>(qw, kw, vw, ow, w1, w2, WQKV, WO, W1P, W2P, 2);
  pack_weights<<<256,  256, 0, stream>>>(qw, kw, vw, ow, w1, w2, WQKV, WO, W1P, W2P, 3);

  // LN1 -> blocked f16 activations
  ln_kernel<<<4096, 256, 0, stream>>>(x, ln1w, ln1b, A16, 1);

  // Fused Q(8 heads)/K/V projection: [32768x256] x [256x2560]
  // waves = (32768/32) * (2560/64) = 40960 -> 5120 blocks of 8 waves
  gemm16<0><<<5120, 256, 0, stream>>>(A16, 256, WQKV, 2560, 256, nullptr, nullptr, QKV, nullptr);

  // Axial attention, both branches; O overwrites Q region of QKV
  attn_kernel<<<dim3(64, 4, 16), 256, 0, stream>>>(QKV);

  // O-projection [32768x2048]x[2048x256] + unblock + x residual -> X1 (f32)
  // waves = 1024 * 4 = 4096 -> 512 blocks
  gemm16<1><<<512, 256, 0, stream>>>(QKV, 2560, WO, 256, 2048, nullptr, x, nullptr, X1);

  // LN2 -> f16
  ln_kernel<<<4096, 256, 0, stream>>>(X1, ln2w, ln2b, H16, 0);

  // MLP: relu(h@w1+b1) -> f16 ; @w2+b2 + X1 -> out (f32)
  gemm16<2><<<512, 256, 0, stream>>>(H16, 256, W1P, 256, 256, b1, nullptr, M16, nullptr);
  gemm16<3><<<512, 256, 0, stream>>>(M16, 256, W2P, 256, 256, b2, X1, nullptr, out);
}
